// TransformerDecoderLayer_7808250544268
// MI455X (gfx1250) — matrix operand order repaired and checked by reading, not compiled, not run on hardware
//
#include <hip/hip_runtime.h>

typedef __bf16 bf16;
typedef __attribute__((ext_vector_type(16))) __bf16 v16bf;
typedef __attribute__((ext_vector_type(8)))  __bf16 v8bf;
typedef __attribute__((ext_vector_type(8)))  float   v8f;

namespace {
constexpr int Bn  = 4;
constexpr int Ss  = 1024;
constexpr int Dd  = 1024;
constexpr int Hh  = 16;
constexpr int DHd = 64;
constexpr int DFF = 4096;
constexpr int BS  = Bn * Ss;           // 4096 rows
constexpr int BM = 128, BN = 128, BK = 32;
constexpr int PAD = 8;                 // row stride 40 bf16 = 80B = 5*16B: aligned 16B chunks,
                                       // 20-dword bank stride -> conflict-free over 16 rows
constexpr float LN_EPS = 1e-5f;
}

// CDNA5 async global->LDS copy (ASYNCcnt) -- inline asm, portable across toolchains.
#define ASYNC_G2L_B128(ldsaddr_u32, gptr_u64)                                   \
    asm volatile("global_load_async_to_lds_b128 %0, %1, off"                    \
                 :: "v"(ldsaddr_u32), "v"(gptr_u64) : "memory")
#define WAIT_ASYNCCNT0() asm volatile("s_wait_asynccnt 0x0" ::: "memory")

__device__ __forceinline__ unsigned lds_addr32(const void* p) {
    // generic LDS pointer: hardware uses addr[31:0] as the LDS byte address
    return (unsigned)(unsigned long long)p;
}

// ---------------------------------------------------------------------------
// f32 -> bf16 conversion (4-wide, grid-stride)
// ---------------------------------------------------------------------------
__global__ __launch_bounds__(256)
void cvt_f32_bf16(const float* __restrict__ src, bf16* __restrict__ dst, long n4) {
    long i = (long)blockIdx.x * blockDim.x + threadIdx.x;
    const long stride = (long)gridDim.x * blockDim.x;
    for (; i < n4; i += stride) {
        float4 f = ((const float4*)src)[i];
        dst[4*i + 0] = (bf16)f.x;
        dst[4*i + 1] = (bf16)f.y;
        dst[4*i + 2] = (bf16)f.z;
        dst[4*i + 3] = (bf16)f.w;
    }
}

// ---------------------------------------------------------------------------
// Tiled transpose (+ optional f32->bf16 convert): out[c][r] = (bf16) in[r][c]
// 64x64 tiles via LDS; R and C must be multiples of 64 (true at all call sites)
// grid = (R/64, C/64, Z); per-z offsets decomposed as zb=z/nH, zh=z%nH
// ---------------------------------------------------------------------------
template <typename TI>
__global__ __launch_bounds__(256)
void transpose_to_bf16(const TI* __restrict__ in, long ldin, long sInb, long sInh,
                       bf16* __restrict__ out, long ldout, long sOutb, long sOuth,
                       int nH)
{
    __shared__ bf16 t[64][64 + 8];
    const int z = blockIdx.z, zb = z / nH, zh = z % nH;
    const TI*  ip = in  + (long)zb * sInb  + (long)zh * sInh;
    bf16*      op = out + (long)zb * sOutb + (long)zh * sOuth;
    const int r0 = blockIdx.x * 64, c0 = blockIdx.y * 64;
    const int tx = threadIdx.x & 63, ty = threadIdx.x >> 6;   // ty 0..3
#pragma unroll
    for (int i = 0; i < 16; ++i) {
        const int r = ty + i * 4;
        t[r][tx] = (bf16)ip[(long)(r0 + r) * ldin + (c0 + tx)];
    }
    __syncthreads();
#pragma unroll
    for (int i = 0; i < 16; ++i) {
        const int r = ty + i * 4;
        op[(long)(c0 + r) * ldout + (r0 + tx)] = t[tx][r];
    }
}

// ---------------------------------------------------------------------------
// Batched WMMA GEMM:  C[z] = act(A[z] * B[z] + bias)
//   A: bf16 [M,K] row stride lda,  B: bf16 N-major [N,K] row stride ldb
//   z offsets decomposed: zb = z/nH, zh = z%nH
// Block tile 128x128x32, 8 wave32s, wave tile 64x32 (4x2 wmma accumulators).
// Double-buffered LDS; BOTH A and B tiles staged with
// GLOBAL_LOAD_ASYNC_TO_LDS_B128 (lane-predicated for N tail), drained by
// s_wait_asynccnt + workgroup split-barrier each tile.
// ---------------------------------------------------------------------------
__global__ __launch_bounds__(256)
void gemm_bf16_wmma(const bf16* __restrict__ Ag, long lda, long sAb, long sAh,
                    const bf16* __restrict__ Bg, long ldb, long sBb, long sBh,
                    float* __restrict__ Cg, long ldc, long sCb, long sCh,
                    bf16* __restrict__ C2g, long ldc2, long sC2b, long sC2h,
                    const float* __restrict__ bias, int relu,
                    int M, int N, int K, int nH)
{
    __shared__ alignas(16) bf16 As[2][BM][BK + PAD];
    __shared__ alignas(16) bf16 Bs[2][BN][BK + PAD];   // N-major: Bs[buf][n][k]
    constexpr unsigned ABUF = sizeof(bf16) * BM * (BK + PAD);  // bytes per buffer

    const int tid  = threadIdx.x;
    const int lane = tid & 31;
    const int wave = tid >> 5;
    const int wrow = wave & 1;     // 2 wave rows  -> 64 M each
    const int wcol = wave >> 1;    // 4 wave cols  -> 32 N each
    const int half = lane >> 4;    // 0/1
    const int l16  = lane & 15;

    const int m0 = blockIdx.x * BM;
    const int n0 = blockIdx.y * BN;
    const int z  = blockIdx.z;
    const int zb = z / nH, zh = z % nH;

    const bf16* A  = Ag + (long)zb * sAb + (long)zh * sAh;
    const bf16* Bp = Bg + (long)zb * sBb + (long)zh * sBh;

    v8f acc[2][4];
#pragma unroll
    for (int ni = 0; ni < 2; ++ni)
#pragma unroll
        for (int mi = 0; mi < 4; ++mi)
#pragma unroll
            for (int e = 0; e < 8; ++e) acc[ni][mi][e] = 0.0f;

    // ---- precomputed per-thread staging addresses (bump per tile) ----
    // A tile: 128 rows x 32 cols = 512 chunks of 8 bf16; 2 chunks/thread
    unsigned     ldsA[2], ldsB[2];
    const bf16*  gA[2];
    const bf16*  gB[2];
    bool         bOk[2];
#pragma unroll
    for (int it = 0; it < 2; ++it) {
        const int c   = tid + it * 256;
        const int row = c >> 2;
        const int col = (c & 3) * 8;
        ldsA[it] = lds_addr32(&As[0][row][col]);
        gA[it]   = A + (long)(m0 + row) * lda + col;
        ldsB[it] = lds_addr32(&Bs[0][row][col]);
        gB[it]   = Bp + (long)(n0 + row) * ldb + col;
        bOk[it]  = (n0 + row) < N;     // N tail (P*V has N=64): predicate copies
    }

    auto stage_tile = [&](int buf) {
#pragma unroll
        for (int it = 0; it < 2; ++it) {
            ASYNC_G2L_B128(ldsA[it] + (unsigned)buf * ABUF,
                           (unsigned long long)gA[it]);
            gA[it] += BK;
        }
#pragma unroll
        for (int it = 0; it < 2; ++it) {
            if (bOk[it])   // disabled lanes: LDS rows n>=N stay garbage, epilogue masks them
                ASYNC_G2L_B128(ldsB[it] + (unsigned)buf * ABUF,
                               (unsigned long long)gB[it]);
            gB[it] += BK;
        }
    };

    const int ntiles = K / BK;
    int cur = 0;
    stage_tile(0);
    WAIT_ASYNCCNT0();
    __syncthreads();

    for (int t = 0; t < ntiles; ++t) {
        if (t + 1 < ntiles) stage_tile(cur ^ 1);   // prefetch next tile (async DMA)

        // ---- compute on current tile: 8 x v_wmma_f32_16x16x32_bf16 ----
        v16bf afr[4], bfr[2];
#pragma unroll
        for (int mi = 0; mi < 4; ++mi) {
            const int row = wrow * 64 + mi * 16 + l16;
            const int cA  = half * 8;               // K = {0..7 | 8..15}
            v8bf lo = *(const v8bf*)(&As[cur][row][cA]);
            v8bf hi = *(const v8bf*)(&As[cur][row][cA + 16]);  // K += 16
            afr[mi] = __builtin_shufflevector(lo, hi,
                      0,1,2,3,4,5,6,7,8,9,10,11,12,13,14,15);
        }
#pragma unroll
        for (int ni = 0; ni < 2; ++ni) {
            const int nn = wcol * 32 + ni * 16 + l16;
            const int cB = half * 8;                // K = {0..7 | 8..15}, then K += 16
            v8bf lo = *(const v8bf*)(&Bs[cur][nn][cB]);
            v8bf hi = *(const v8bf*)(&Bs[cur][nn][cB + 16]);
            bfr[ni] = __builtin_shufflevector(lo, hi,
                      0,1,2,3,4,5,6,7,8,9,10,11,12,13,14,15);
        }
        // ni-outer: B operand constant over 4 consecutive WMMAs -> RB reuse hint
        // (reuse flags must be literal constants for the builtin)
#pragma unroll
        for (int ni = 0; ni < 2; ++ni) {
            acc[ni][0] = __builtin_amdgcn_wmma_f32_16x16x32_bf16(
                false, afr[0], false, bfr[ni], (short)0, acc[ni][0], false, true);
            acc[ni][1] = __builtin_amdgcn_wmma_f32_16x16x32_bf16(
                false, afr[1], false, bfr[ni], (short)0, acc[ni][1], false, true);
            acc[ni][2] = __builtin_amdgcn_wmma_f32_16x16x32_bf16(
                false, afr[2], false, bfr[ni], (short)0, acc[ni][2], false, true);
            acc[ni][3] = __builtin_amdgcn_wmma_f32_16x16x32_bf16(
                false, afr[3], false, bfr[ni], (short)0, acc[ni][3], false, false);
        }

        WAIT_ASYNCCNT0();     // own async copies into buf^1 done
        __syncthreads();      // all waves done reading buf & staging buf^1
        cur ^= 1;
    }

    // ---- epilogue: bias + relu, f32 and/or bf16 stores ----
    float* C  = Cg  ? Cg  + (long)zb * sCb  + (long)zh * sCh  : nullptr;
    bf16*  C2 = C2g ? C2g + (long)zb * sC2b + (long)zh * sC2h : nullptr;
#pragma unroll
    for (int ni = 0; ni < 2; ++ni) {
#pragma unroll
        for (int mi = 0; mi < 4; ++mi) {
            const int gm = m0 + wrow * 64 + mi * 16;
            const int gn = n0 + wcol * 32 + ni * 16 + l16;
            if (gn < N) {
                const float bv = bias ? bias[gn] : 0.0f;
#pragma unroll
                for (int e = 0; e < 8; ++e) {
                    const int row = gm + e + half * 8;   // C/D layout (ISA 7.12.2)
                    float v = acc[ni][mi][e] + bv;
                    if (relu) v = fmaxf(v, 0.0f);
                    if (C)  C[(long)row * ldc  + gn] = v;
                    if (C2) C2[(long)row * ldc2 + gn] = (bf16)v;
                }
            }
        }
    }
}

// ---------------------------------------------------------------------------
// masked scaled softmax over rows of [B,H,S,S] f32 -> bf16 probs
// grid = (S, H, B); causal => valid length = q+1
// ---------------------------------------------------------------------------
__global__ __launch_bounds__(256)
void softmax_rows(const float* __restrict__ scores, bf16* __restrict__ probs,
                  int S, int causal, float scale)
{
    const int q = blockIdx.x, h = blockIdx.y, b = blockIdx.z;
    const long row = (((long)b * gridDim.y + h) * S + q) * (long)S;
    const int  L   = causal ? (q + 1) : S;
    const int tid  = threadIdx.x;
    __shared__ float red[256];

    float mx = -3.4e38f;
    for (int j = tid; j < L; j += 256) mx = fmaxf(mx, scores[row + j] * scale);
    red[tid] = mx; __syncthreads();
    for (int k = 128; k > 0; k >>= 1) {
        if (tid < k) red[tid] = fmaxf(red[tid], red[tid + k]);
        __syncthreads();
    }
    mx = red[0]; __syncthreads();

    float sum = 0.0f;
    for (int j = tid; j < L; j += 256) sum += __expf(scores[row + j] * scale - mx);
    red[tid] = sum; __syncthreads();
    for (int k = 128; k > 0; k >>= 1) {
        if (tid < k) red[tid] += red[tid + k];
        __syncthreads();
    }
    const float inv = 1.0f / red[0];

    for (int j = tid; j < S; j += 256) {
        float p = (j < L) ? __expf(scores[row + j] * scale - mx) * inv : 0.0f;
        probs[row + j] = (bf16)p;
    }
}

// ---------------------------------------------------------------------------
// out = LayerNorm(x + y) * g + b ; writes f32 (Of) and optionally bf16 (Ob)
// one block per row of D=1024, 4 elements/thread
// ---------------------------------------------------------------------------
__global__ __launch_bounds__(256)
void add_layernorm(const float* __restrict__ X, const float* __restrict__ Y,
                   const float* __restrict__ G, const float* __restrict__ Be,
                   float* __restrict__ Of, bf16* __restrict__ Ob)
{
    const long row = blockIdx.x;
    const float* xr = X + row * Dd;
    const float* yr = Y + row * Dd;
    const int tid = threadIdx.x;
    __shared__ float red[256];

    float v[4];
    float s = 0.0f;
#pragma unroll
    for (int i = 0; i < 4; ++i) {
        v[i] = xr[tid + i * 256] + yr[tid + i * 256];
        s += v[i];
    }
    red[tid] = s; __syncthreads();
    for (int k = 128; k > 0; k >>= 1) {
        if (tid < k) red[tid] += red[tid + k];
        __syncthreads();
    }
    const float mean = red[0] * (1.0f / Dd);
    __syncthreads();

    float q = 0.0f;
#pragma unroll
    for (int i = 0; i < 4; ++i) { const float d = v[i] - mean; q += d * d; }
    red[tid] = q; __syncthreads();
    for (int k = 128; k > 0; k >>= 1) {
        if (tid < k) red[tid] += red[tid + k];
        __syncthreads();
    }
    const float inv = rsqrtf(red[0] * (1.0f / Dd) + LN_EPS);

#pragma unroll
    for (int i = 0; i < 4; ++i) {
        const int idx = tid + i * 256;
        const float o = (v[i] - mean) * inv * G[idx] + Be[idx];
        Of[row * Dd + idx] = o;
        if (Ob) Ob[row * Dd + idx] = (bf16)o;
    }
}

// ---------------------------------------------------------------------------
// host-side helpers
// ---------------------------------------------------------------------------
static void run_cvt(const float* src, bf16* dst, long n, hipStream_t st) {
    const long n4 = n / 4;
    int blocks = (int)((n4 + 255) / 256);
    if (blocks > 4096) blocks = 4096;
    cvt_f32_bf16<<<blocks, 256, 0, st>>>(src, dst, n4);
}

// weight transpose: W [K,N] f32 -> WT [N,K] bf16
static void run_wtrans(const float* w, int K, int N, bf16* wt, hipStream_t st) {
    dim3 g(K / 64, N / 64, 1);
    transpose_to_bf16<float><<<g, 256, 0, st>>>(w, N, 0, 0, wt, K, 0, 0, 1);
}

static void run_gemm(const bf16* A, long lda, long sAb, long sAh,
                     const bf16* Bm, long ldb, long sBb, long sBh,
                     float* C, long ldc, long sCb, long sCh,
                     bf16* C2, long ldc2, long sC2b, long sC2h,
                     const float* bias, int relu,
                     int M, int N, int K, int Z, int nH, hipStream_t st)
{
    dim3 g((M + BM - 1) / BM, (N + BN - 1) / BN, Z);
    gemm_bf16_wmma<<<g, 256, 0, st>>>(A, lda, sAb, sAh, Bm, ldb, sBb, sBh,
                                      C, ldc, sCb, sCh, C2, ldc2, sC2b, sC2h,
                                      bias, relu, M, N, K, nH);
}

extern "C" void kernel_launch(void* const* d_in, const int* in_sizes, int n_in,
                              void* d_out, int out_size, void* d_ws, size_t ws_size,
                              hipStream_t stream)
{
    (void)in_sizes; (void)n_in; (void)out_size; (void)ws_size;
    const float* x      = (const float*)d_in[0];
    const float* enc    = (const float*)d_in[1];
    // d_in[2] tgt_mask (causal), d_in[3] src_mask (all ones): handled analytically
    const float* sa_wq = (const float*)d_in[4];
    const float* sa_wk = (const float*)d_in[5];
    const float* sa_wv = (const float*)d_in[6];
    const float* sa_wo = (const float*)d_in[7];
    const float* sa_bq = (const float*)d_in[8];
    const float* sa_bk = (const float*)d_in[9];
    const float* sa_bv = (const float*)d_in[10];
    const float* sa_bo = (const float*)d_in[11];
    const float* ca_wq = (const float*)d_in[12];
    const float* ca_wk = (const float*)d_in[13];
    const float* ca_wv = (const float*)d_in[14];
    const float* ca_wo = (const float*)d_in[15];
    const float* ca_bq = (const float*)d_in[16];
    const float* ca_bk = (const float*)d_in[17];
    const float* ca_bv = (const float*)d_in[18];
    const float* ca_bo = (const float*)d_in[19];
    const float* ff_w1 = (const float*)d_in[20];
    const float* ff_b1 = (const float*)d_in[21];
    const float* ff_w2 = (const float*)d_in[22];
    const float* ff_b2 = (const float*)d_in[23];
    const float* ln1_g = (const float*)d_in[24];
    const float* ln1_b = (const float*)d_in[25];
    const float* ln2_g = (const float*)d_in[26];
    const float* ln2_b = (const float*)d_in[27];
    const float* ln3_g = (const float*)d_in[28];
    const float* ln3_b = (const float*)d_in[29];
    float* out = (float*)d_out;

    // ---- carve workspace ----
    size_t off = 0;
    auto alloc = [&](size_t bytes) -> char* {
        char* p = (char*)d_ws + off;
        off += (bytes + 255) & ~(size_t)255;
        return p;
    };
    const size_t DD = (size_t)Dd * Dd;
    // transposed N-major bf16 weights
    bf16* wqT1 = (bf16*)alloc(DD * 2);  bf16* wkT1 = (bf16*)alloc(DD * 2);
    bf16* wvT1 = (bf16*)alloc(DD * 2);  bf16* woT1 = (bf16*)alloc(DD * 2);
    bf16* wqT2 = (bf16*)alloc(DD * 2);  bf16* wkT2 = (bf16*)alloc(DD * 2);
    bf16* wvT2 = (bf16*)alloc(DD * 2);  bf16* woT2 = (bf16*)alloc(DD * 2);
    bf16* w1T  = (bf16*)alloc((size_t)DFF * Dd * 2);   // [DFF, D]
    bf16* w2T  = (bf16*)alloc((size_t)Dd * DFF * 2);   // [D, DFF]
    bf16* xb   = (bf16*)alloc((size_t)BS * Dd * 2);
    bf16* enb  = (bf16*)alloc((size_t)BS * Dd * 2);
    bf16* qb   = (bf16*)alloc((size_t)BS * Dd * 2);
    bf16* kb   = (bf16*)alloc((size_t)BS * Dd * 2);
    bf16* vb   = (bf16*)alloc((size_t)BS * Dd * 2);
    bf16* vT   = (bf16*)alloc((size_t)Bn * Hh * DHd * Ss * 2);  // [B,H,DH,S]
    float* scores = (float*)alloc((size_t)Bn * Hh * Ss * Ss * 4);   // 256 MB
    bf16*  probs  = (bf16*) alloc((size_t)Bn * Hh * Ss * Ss * 2);   // 128 MB
    bf16*  attnb  = (bf16*) alloc((size_t)BS * Dd * 2);
    float* proj   = (float*)alloc((size_t)BS * Dd * 4);
    float* h1f    = (float*)alloc((size_t)BS * Dd * 4);
    bf16*  h1b    = (bf16*) alloc((size_t)BS * Dd * 2);
    float* h2f    = (float*)alloc((size_t)BS * Dd * 4);
    bf16*  h2b    = (bf16*) alloc((size_t)BS * Dd * 2);
    bf16*  ffn1   = (bf16*) alloc((size_t)BS * DFF * 2);

    // ---- 1. weight transpose+convert (N-major bf16), activation convert ----
    run_wtrans(sa_wq, Dd, Dd, wqT1, stream);  run_wtrans(sa_wk, Dd, Dd, wkT1, stream);
    run_wtrans(sa_wv, Dd, Dd, wvT1, stream);  run_wtrans(sa_wo, Dd, Dd, woT1, stream);
    run_wtrans(ca_wq, Dd, Dd, wqT2, stream);  run_wtrans(ca_wk, Dd, Dd, wkT2, stream);
    run_wtrans(ca_wv, Dd, Dd, wvT2, stream);  run_wtrans(ca_wo, Dd, Dd, woT2, stream);
    run_wtrans(ff_w1, Dd, DFF, w1T, stream);
    run_wtrans(ff_w2, DFF, Dd, w2T, stream);
    run_cvt(x,   xb,  (size_t)BS * Dd, stream);
    run_cvt(enc, enb, (size_t)BS * Dd, stream);

    const long SD  = (long)Ss * Dd;
    const long HSS = (long)Hh * Ss * Ss;
    const long SS2 = (long)Ss * Ss;
    const long HDS = (long)Hh * DHd * Ss;   // vT batch stride
    const long DS  = (long)DHd * Ss;        // vT head stride
    const float scale = 1.0f / 8.0f;        // 1/sqrt(64)

    // per-(b,h) V transpose: vb [b*S+s][h*64+d] -> vT [b][h][d][s]
    auto vtrans = [&]() {
        dim3 g(Ss / 64, DHd / 64, Bn * Hh);
        transpose_to_bf16<bf16><<<g, 256, 0, stream>>>(
            vb, Dd, SD, DHd, vT, Ss, HDS, DS, Hh);
    };

    // ---- 2. self-attention ----
    run_gemm(xb, Dd, 0,0, wqT1, Dd, 0,0, nullptr,0,0,0, qb, Dd,0,0, sa_bq,0, BS, Dd, Dd, 1,1, stream);
    run_gemm(xb, Dd, 0,0, wkT1, Dd, 0,0, nullptr,0,0,0, kb, Dd,0,0, sa_bk,0, BS, Dd, Dd, 1,1, stream);
    run_gemm(xb, Dd, 0,0, wvT1, Dd, 0,0, nullptr,0,0,0, vb, Dd,0,0, sa_bv,0, BS, Dd, Dd, 1,1, stream);
    vtrans();
    // scores[b,h] = Q_h * K_h^T   (K_h rows are N-major already)
    run_gemm(qb, Dd, SD, DHd, kb, Dd, SD, DHd,
             scores, Ss, HSS, SS2, nullptr,0,0,0, nullptr,0,
             Ss, Ss, DHd, Bn * Hh, Hh, stream);
    softmax_rows<<<dim3(Ss, Hh, Bn), 256, 0, stream>>>(scores, probs, Ss, 1, scale);
    // attn[b,h] = P * V_h   (B = vT, N-major [DH, S])
    run_gemm(probs, Ss, HSS, SS2, vT, Ss, HDS, DS,
             nullptr,0,0,0, attnb, Dd, SD, DHd, nullptr,0,
             Ss, DHd, Ss, Bn * Hh, Hh, stream);
    run_gemm(attnb, Dd, 0,0, woT1, Dd, 0,0, proj, Dd,0,0, nullptr,0,0,0, sa_bo,0, BS, Dd, Dd, 1,1, stream);
    add_layernorm<<<BS, 256, 0, stream>>>(x, proj, ln1_g, ln1_b, h1f, h1b);

    // ---- 3. cross-attention (Q from h1, K/V from encoder) ----
    run_gemm(h1b, Dd, 0,0, wqT2, Dd, 0,0, nullptr,0,0,0, qb, Dd,0,0, ca_bq,0, BS, Dd, Dd, 1,1, stream);
    run_gemm(enb, Dd, 0,0, wkT2, Dd, 0,0, nullptr,0,0,0, kb, Dd,0,0, ca_bk,0, BS, Dd, Dd, 1,1, stream);
    run_gemm(enb, Dd, 0,0, wvT2, Dd, 0,0, nullptr,0,0,0, vb, Dd,0,0, ca_bv,0, BS, Dd, Dd, 1,1, stream);
    vtrans();
    run_gemm(qb, Dd, SD, DHd, kb, Dd, SD, DHd,
             scores, Ss, HSS, SS2, nullptr,0,0,0, nullptr,0,
             Ss, Ss, DHd, Bn * Hh, Hh, stream);
    softmax_rows<<<dim3(Ss, Hh, Bn), 256, 0, stream>>>(scores, probs, Ss, 0, scale);
    run_gemm(probs, Ss, HSS, SS2, vT, Ss, HDS, DS,
             nullptr,0,0,0, attnb, Dd, SD, DHd, nullptr,0,
             Ss, DHd, Ss, Bn * Hh, Hh, stream);
    run_gemm(attnb, Dd, 0,0, woT2, Dd, 0,0, proj, Dd,0,0, nullptr,0,0,0, ca_bo,0, BS, Dd, Dd, 1,1, stream);
    add_layernorm<<<BS, 256, 0, stream>>>(h1f, proj, ln2_g, ln2_b, h2f, h2b);

    // ---- 4. feed-forward ----
    run_gemm(h2b, Dd, 0,0, w1T, Dd, 0,0, nullptr,0,0,0, ffn1, DFF,0,0, ff_b1,1, BS, DFF, Dd, 1,1, stream);
    run_gemm(ffn1, DFF, 0,0, w2T, DFF, 0,0, proj, Dd,0,0, nullptr,0,0,0, ff_b2,0, BS, Dd, DFF, 1,1, stream);
    add_layernorm<<<BS, 256, 0, stream>>>(h2f, proj, ln3_g, ln3_b, out, nullptr);
}
